// GRACE_3934190043973
// MI455X (gfx1250) — matrix operand order repaired and checked by reading, not compiled, not run on hardware
//
#include <hip/hip_runtime.h>

#define N_NODES 100000
#define N_EDGES 1600000
#define F_INF   128     // input features / hidden width
#define F_HID   128     // layer-1 output width
#define F_OUT   64      // layer-2 output width

typedef __attribute__((ext_vector_type(16))) _Float16 v16h;
typedef __attribute__((ext_vector_type(8)))  _Float16 v8h;
typedef __attribute__((ext_vector_type(8)))  float    v8f;

// ---------------------------------------------------------------- degree/norm
__global__ void k_init_deg(unsigned* __restrict__ deg, int n) {
    int i = blockIdx.x * blockDim.x + threadIdx.x;
    if (i < n) deg[i] = 1u;                         // self-loop contributes 1
}

__global__ void k_count_deg(const long long* __restrict__ col,
                            unsigned* __restrict__ deg, int e) {
    int i = blockIdx.x * blockDim.x + threadIdx.x;
    if (i < e) atomicAdd(&deg[(int)col[i]], 1u);
}

__global__ void k_dinv(const unsigned* __restrict__ deg,
                       float* __restrict__ dinv, int n) {
    int i = blockIdx.x * blockDim.x + threadIdx.x;
    if (i < n) dinv[i] = rsqrtf((float)deg[i]);     // deg >= 1 always
}

// ---------------------------------------------------------------- conversions
__global__ void k_f32_to_f16(const float* __restrict__ src,
                             _Float16* __restrict__ dst, int n) {
    int i = blockIdx.x * blockDim.x + threadIdx.x;
    if (i < n) dst[i] = (_Float16)src[i];
}

// Pack W [K, OUT] (f32, row-major) into per-lane WMMA B fragments (f16).
// Fragment index: t = ((nt*(K/32) + ks)*32 + lane)*16 + e
// B layout (16-bit, 32x16): lanes 0-15 -> n = lane, K = ks*32 + e (e=0..15);
//                           lanes 16-31 -> K = ks*32 + 16 + e.
__global__ void k_pack_w(const float* __restrict__ W, _Float16* __restrict__ wf,
                         int K, int OUT) {
    int t = blockIdx.x * blockDim.x + threadIdx.x;
    if (t >= K * OUT) return;
    int e  = t & 15;
    int L  = (t >> 4) & 31;
    int ks = (t >> 9) % (K >> 5);
    int nt = t / ((K >> 5) << 9);
    int n  = nt * 16 + (L & 15);
    int k  = ks * 32 + ((L >> 4) << 4) + e;
    wf[t] = (_Float16)W[(size_t)k * OUT + n];
}

// ---------------------------------------------------------------- WMMA GEMM
// hs[m,n] = acc[m,n] = dinv[m] * sum_k A[m,k] * W[k,n]
// One wave computes one 16x16 tile; K/32 v_wmma_f32_16x16x32_f16 per tile.
template<int K, int OUT>
__global__ void k_gemm_scaled(const _Float16* __restrict__ A,
                              const _Float16* __restrict__ WF,
                              const float* __restrict__ dinv,
                              float* __restrict__ hs,
                              float* __restrict__ acc,
                              int numTiles) {
    int wave = blockIdx.x * (blockDim.x >> 5) + (threadIdx.x >> 5);
    if (wave >= numTiles) return;                   // wave-uniform: EXEC stays all-1s
    const int NT = OUT / 16;
    int mt   = wave / NT;
    int nt   = wave % NT;
    int lane = threadIdx.x & 31;
    int mrow = mt * 16 + (lane & 15);               // A: lanes 0-15 & 16-31 both M=0..15
    int khalf = lane >> 4;                          // selects 16-wide K half (contiguous k order, matches B packing)

    const _Float16* arow = A + (size_t)mrow * K;
    const _Float16* wfp  = WF + (size_t)nt * (K / 32) * 32 * 16;

    v8f c = {};
    #pragma unroll
    for (int ks = 0; ks < K / 32; ++ks) {
        // A fragment: elems 0..15 = K[kb + khalf*16 .. +15]
        v8h alo = *(const v8h*)(arow + ks * 32 + khalf * 16);
        v8h ahi = *(const v8h*)(arow + ks * 32 + khalf * 16 + 8);
        // B fragment: 16 contiguous packed halves per lane
        const _Float16* wl = wfp + ((size_t)ks * 32 + lane) * 16;
        v8h blo = *(const v8h*)(wl);
        v8h bhi = *(const v8h*)(wl + 8);
        v16h av, bv;
        #pragma unroll
        for (int i = 0; i < 8; ++i) {
            av[i] = alo[i]; av[8 + i] = ahi[i];
            bv[i] = blo[i]; bv[8 + i] = bhi[i];
        }
        c = __builtin_amdgcn_wmma_f32_16x16x32_f16(false, av, false, bv,
                                                   (short)0, c, false, false);
    }

    // C/D layout: elem j -> M = mt*16 + (lane>>4)*8 + j, N = nt*16 + (lane&15)
    #pragma unroll
    for (int j = 0; j < 8; ++j) {
        int m = mt * 16 + khalf * 8 + j;
        float v = c[j] * dinv[m];
        size_t o = (size_t)m * OUT + nt * 16 + (lane & 15);
        hs[o]  = v;
        acc[o] = v;       // self-loop term: acc starts at hs (dinv[c]*h[c])
    }
}

// ---------------------------------------------------------------- edge scatter
// One wave per edge: gather F contiguous floats of hs[row], f32-atomic-add into acc[col].
template<int F, int VEC>
__global__ void k_scatter(const float* __restrict__ hs,
                          const long long* __restrict__ rows,
                          const long long* __restrict__ cols,
                          float* __restrict__ acc, int e) {
    int w = blockIdx.x * (blockDim.x >> 5) + (threadIdx.x >> 5);
    int lane = threadIdx.x & 31;
    if (w >= e) return;
    int r = __builtin_amdgcn_readfirstlane((int)rows[w]);
    int c = __builtin_amdgcn_readfirstlane((int)cols[w]);
    const float* src = hs  + (size_t)r * F + lane * VEC;
    float*       dst = acc + (size_t)c * F + lane * VEC;
    float v[VEC];
    #pragma unroll
    for (int i = 0; i < VEC; ++i) v[i] = src[i];    // coalesced b128/b64 load
    #pragma unroll
    for (int i = 0; i < VEC; ++i) unsafeAtomicAdd(dst + i, v[i]);  // global_atomic_add_f32
}

// ---------------------------------------------------------------- epilogues
__global__ void k_finalize_h16(const float* __restrict__ acc,
                               const float* __restrict__ dinv,
                               const float* __restrict__ b,
                               _Float16* __restrict__ h16, int total, int F) {
    int i = blockIdx.x * blockDim.x + threadIdx.x;
    if (i >= total) return;
    int cnode = i / F, j = i % F;
    float v = fmaf(dinv[cnode], acc[i], b[j]);
    h16[i] = (_Float16)(v > 0.0f ? v : 0.0f);
}

__global__ void k_finalize_out(const float* __restrict__ acc,
                               const float* __restrict__ dinv,
                               const float* __restrict__ b,
                               float* __restrict__ z, int total, int F) {
    int i = blockIdx.x * blockDim.x + threadIdx.x;
    if (i >= total) return;
    int cnode = i / F, j = i % F;
    float v = fmaf(dinv[cnode], acc[i], b[j]);
    z[i] = v > 0.0f ? v : 0.0f;
}

// ---------------------------------------------------------------- launch
extern "C" void kernel_launch(void* const* d_in, const int* in_sizes, int n_in,
                              void* d_out, int out_size, void* d_ws, size_t ws_size,
                              hipStream_t stream) {
    const float*     x   = (const float*)d_in[0];
    const long long* ei  = (const long long*)d_in[1];   // [2, E] int64
    const float*     W1  = (const float*)d_in[2];
    const float*     b1  = (const float*)d_in[3];
    const float*     W2  = (const float*)d_in[4];
    const float*     b2  = (const float*)d_in[5];
    const long long* erow = ei;
    const long long* ecol = ei + N_EDGES;
    float* zout = (float*)d_out;

    // ---- workspace carve-up (256B aligned)
    char* ws = (char*)d_ws;
    size_t off = 0;
    auto carve = [&](size_t bytes) -> char* {
        char* p = ws + off;
        off = (off + bytes + 255) & ~(size_t)255;
        return p;
    };
    unsigned* deg  = (unsigned*)carve((size_t)N_NODES * 4);
    float*    dinv = (float*)   carve((size_t)N_NODES * 4);
    _Float16* a16  = (_Float16*)carve((size_t)N_NODES * F_INF * 2);  // x16, reused as h16
    _Float16* wf1  = (_Float16*)carve((size_t)F_INF * F_HID * 2);
    _Float16* wf2  = (_Float16*)carve((size_t)F_HID * F_OUT * 2);
    float*    hs   = (float*)   carve((size_t)N_NODES * F_HID * 4);  // reused for layer 2
    float*    acc  = (float*)   carve((size_t)N_NODES * F_HID * 4);  // reused for layer 2

    const int T = 256;
    auto nb = [](long long n, int t) { return (int)((n + t - 1) / t); };

    // ---- normalization factors
    k_init_deg <<<nb(N_NODES, T), T, 0, stream>>>(deg, N_NODES);
    k_count_deg<<<nb(N_EDGES, T), T, 0, stream>>>(ecol, deg, N_EDGES);
    k_dinv     <<<nb(N_NODES, T), T, 0, stream>>>(deg, dinv, N_NODES);

    // ---- precision convert + weight packing
    k_f32_to_f16<<<nb((long long)N_NODES * F_INF, T), T, 0, stream>>>(x, a16, N_NODES * F_INF);
    k_pack_w<<<nb(F_INF * F_HID, T), T, 0, stream>>>(W1, wf1, F_INF, F_HID);
    k_pack_w<<<nb(F_HID * F_OUT, T), T, 0, stream>>>(W2, wf2, F_HID, F_OUT);

    // ---- layer 1: GEMM (dinv-scaled) -> scatter -> relu/bias -> h16
    {
        int tiles = (N_NODES / 16) * (F_HID / 16);          // 50000
        k_gemm_scaled<F_INF, F_HID><<<nb(tiles, 8), T, 0, stream>>>(a16, wf1, dinv, hs, acc, tiles);
        k_scatter<F_HID, 4><<<nb((long long)N_EDGES * 32, T), T, 0, stream>>>(hs, erow, ecol, acc, N_EDGES);
        k_finalize_h16<<<nb((long long)N_NODES * F_HID, T), T, 0, stream>>>(acc, dinv, b1, a16,
                                                                            N_NODES * F_HID, F_HID);
    }

    // ---- layer 2: GEMM (dinv-scaled) -> scatter -> relu/bias -> out
    {
        int tiles = (N_NODES / 16) * (F_OUT / 16);          // 25000
        k_gemm_scaled<F_HID, F_OUT><<<nb(tiles, 8), T, 0, stream>>>(a16, wf2, dinv, hs, acc, tiles);
        k_scatter<F_OUT, 2><<<nb((long long)N_EDGES * 32, T), T, 0, stream>>>(hs, erow, ecol, acc, N_EDGES);
        k_finalize_out<<<nb((long long)N_NODES * F_OUT, T), T, 0, stream>>>(acc, dinv, b2, zout,
                                                                            N_NODES * F_OUT, F_OUT);
    }
}
